// PotentialNet_50663434224358
// MI455X (gfx1250) — matrix operand order repaired and checked by reading, not compiled, not run on hardware
//
#include <hip/hip_runtime.h>
#include <hip/hip_bf16.h>
#include <cmath>

typedef __attribute__((ext_vector_type(16))) _Float16 v16h;
typedef __attribute__((ext_vector_type(8)))  _Float16 v8h;
typedef __attribute__((ext_vector_type(8)))  float    v8f;

#define DEV __device__ __forceinline__

DEV float sigmoid_f(float x) { return 1.0f / (1.0f + __expf(-x)); }

DEV v16h cat8(v8h a, v8h b) {
  return __builtin_shufflevector(a, b, 0, 1, 2, 3, 4, 5, 6, 7, 8, 9, 10, 11, 12,
                                 13, 14, 15);
}

// ---- fragment loaders: straight-line aligned vector loads, no bounds checks ----
// A fragment (16x32 f16): lane half hi: blocks [k0+hi*8, +8) and [k0+16+hi*8, +8)
DEV v16h load_a16(const _Float16* A, int lda, int m0, int k0) {
  const int lane = threadIdx.x & 31;
  const int hi = lane >> 4;
  const int m = m0 + (lane & 15);
  const _Float16* p = A + (size_t)m * lda + k0 + hi * 8;
  return cat8(*(const v8h*)p, *(const v8h*)(p + 16));
}

// A fragment from f32 source (convert on load)
DEV v16h load_a32(const float* A, int lda, int m0, int k0) {
  const int lane = threadIdx.x & 31;
  const int hi = lane >> 4;
  const int m = m0 + (lane & 15);
  const float* p = A + (size_t)m * lda + k0 + hi * 8;
  v8f lo = *(const v8f*)p;
  v8f hb = *(const v8f*)(p + 16);
  v16h f;
#pragma unroll
  for (int i = 0; i < 8; ++i) {
    f[i] = (_Float16)lo[i];
    f[i + 8] = (_Float16)hb[i];
  }
  return f;
}

// A fragment from logical concat [H(0..Fh) | X(Fh..)] ; Fh multiple of 8
DEV v16h load_a16_cat(const _Float16* H, int ldh, int Fh, const _Float16* X,
                      int ldx, int m0, int k0) {
  const int lane = threadIdx.x & 31;
  const int hi = lane >> 4;
  const int m = m0 + (lane & 15);
  const int b0 = k0 + hi * 8;
  const int b1 = k0 + 16 + hi * 8;
  const _Float16* p0 =
      (b0 < Fh) ? (H + (size_t)m * ldh + b0) : (X + (size_t)m * ldx + (b0 - Fh));
  const _Float16* p1 =
      (b1 < Fh) ? (H + (size_t)m * ldh + b1) : (X + (size_t)m * ldx + (b1 - Fh));
  return cat8(*(const v8h*)p0, *(const v8h*)p1);
}

// B fragment (32x16 f16) from pre-packed BT layout: Bp[n][k], ldk mult of 32
DEV v16h load_b16(const _Float16* Bp, int ldk, int n0, int k0) {
  const int lane = threadIdx.x & 31;
  const int hi = lane >> 4;
  const int n = n0 + (lane & 15);
  const _Float16* p = Bp + (size_t)n * ldk + k0 + hi * 8;
  return cat8(*(const v8h*)p, *(const v8h*)(p + 16));
}

DEV v8f wmma16(v16h a, v16h b, v8f c) {
  return __builtin_amdgcn_wmma_f32_16x16x32_f16(false, a, false, b, (short)0, c,
                                                false, false);
}

// ---------------- utility kernels ----------------

__global__ void k_zero(float* p, long long n) {
  long long i = (long long)blockIdx.x * blockDim.x + threadIdx.x;
  const long long st = (long long)gridDim.x * blockDim.x;
  for (; i < n; i += st) p[i] = 0.0f;
}

// f32 [N x Fx] -> f16 [Npad x Fo] with zero padding (rows >= N zeroed too)
__global__ void k_cvtpad(const float* X, int Nn, int Fx, _Float16* out, int Fo,
                         int Npad) {
  long long i = (long long)blockIdx.x * blockDim.x + threadIdx.x;
  const long long tot = (long long)Npad * Fo;
  const long long st = (long long)gridDim.x * blockDim.x;
  for (; i < tot; i += st) {
    const int f = (int)(i % Fo);
    const long long v = i / Fo;
    float val = 0.0f;
    if (v < Nn && f < Fx) val = X[v * Fx + f];
    out[i] = (_Float16)val;
  }
}

__global__ void k_copy16(const _Float16* src, _Float16* dst, long long n) {
  long long i = (long long)blockIdx.x * blockDim.x + threadIdx.x;
  const long long st = (long long)gridDim.x * blockDim.x;
  for (; i < n; i += st) dst[i] = src[i];
}

// pack W [rows x K] f32 -> [rows x Kpad] f16 (zero pad)
__global__ void k_packB(const float* W, int rows, int K, _Float16* out, int Kpad) {
  int i = blockIdx.x * blockDim.x + threadIdx.x;
  const int tot = rows * Kpad;
  const int st = gridDim.x * blockDim.x;
  for (; i < tot; i += st) {
    const int k = i % Kpad, r = i / Kpad;
    out[i] = (_Float16)((k < K) ? W[r * K + k] : 0.0f);
  }
}

// pack concatenated etype weights: out[n][k] = Wet[t= k/F][n][kk=k%F], n<F
__global__ void k_packEt(const float* Wet, int F, int NE, _Float16* out, int Kpad) {
  int i = blockIdx.x * blockDim.x + threadIdx.x;
  const int tot = F * Kpad;
  const int st = gridDim.x * blockDim.x;
  for (; i < tot; i += st) {
    const int k = i % Kpad, n = i / Kpad;
    float v = 0.0f;
    if (k < NE * F) {
      const int t = k / F, kk = k - t * F;
      v = Wet[((long long)t * F + n) * F + kk];
    }
    out[i] = (_Float16)v;
  }
}

// ---------------- graph stage kernels ----------------

// per-edge: s[dst, etype, :] += h[src, :] ; cnt[dst, etype] += 1  (h is f16, ld 64)
__global__ void k_scatter(const _Float16* H, int F, const int* src, const int* dst,
                          const int* et, int E, int NE, float* S, float* cnt) {
  const int e = blockIdx.x * blockDim.x + threadIdx.x;
  if (e >= E) return;
  const int t = et[e];
  const _Float16* hr = H + (size_t)src[e] * 64;
  float* sr = S + ((long long)dst[e] * NE + t) * F;
  if (e + 2048 < E) __builtin_prefetch(H + (size_t)src[e + 2048] * 64, 0, 1);
  for (int f = 0; f < F; f += 8) {
    v8h hv = *(const v8h*)(hr + f);
#pragma unroll
    for (int j = 0; j < 8; ++j) atomicAdd(&sr[f + j], (float)hv[j]);
  }
  atomicAdd(&cnt[(long long)dst[e] * NE + t], 1.0f);
}

// a[v,:] = sum_t s[v,t,:] @ W[t]^T + cnt[v,t]*b[t]; S f32 [Npad x 256], etP f16
__global__ void k_msg(const float* S, const float* cnt, const _Float16* etP,
                      const float* bias, _Float16* Ao, int Nn, int Npad, int F,
                      int NE) {
  const int wid = threadIdx.x >> 5;
  const int m0 = (blockIdx.x * (blockDim.x >> 5) + wid) * 16;
  if (m0 >= Npad) return;
  const int n0 = blockIdx.y * 16;
  v8f c = {0.f, 0.f, 0.f, 0.f, 0.f, 0.f, 0.f, 0.f};
  for (int k0 = 0; k0 < 256; k0 += 32) {
    v16h af = load_a32(S, 256, m0, k0);
    v16h bf = load_b16(etP, 256, n0, k0);
    c = wmma16(af, bf, c);
  }
  const int lane = threadIdx.x & 31;
  const int hi = lane >> 4;
  const int n = n0 + (lane & 15);
#pragma unroll
  for (int r = 0; r < 8; ++r) {
    const int m = m0 + r + 8 * hi;
    if (m < Nn) {
      float acc = c[r];
      for (int t = 0; t < NE; ++t)
        acc += cnt[(long long)m * NE + t] * bias[t * F + n];
      Ao[(size_t)m * 64 + n] = (_Float16)acc;
    }
  }
}

// fused GRU cell: 6 WMMA accumulators, K loop fixed 64 (zero-padded)
__global__ void k_gru(const _Float16* A, const _Float16* H, const _Float16* wiP,
                      const _Float16* whP, const float* bi, const float* bh,
                      _Float16* Hn, int Nn, int Npad, int F) {
  const int wid = threadIdx.x >> 5;
  const int m0 = (blockIdx.x * (blockDim.x >> 5) + wid) * 16;
  if (m0 >= Npad) return;
  const int n0 = blockIdx.y * 16;
  v8f cir = {0.f, 0.f, 0.f, 0.f, 0.f, 0.f, 0.f, 0.f};
  v8f ciz = cir, cin = cir, chr_ = cir, chz = cir, chn = cir;
#pragma unroll
  for (int k0 = 0; k0 < 64; k0 += 32) {
    v16h af = load_a16(A, 64, m0, k0);
    v16h hf = load_a16(H, 64, m0, k0);
    cir = wmma16(af, load_b16(wiP, 64, n0, k0), cir);
    ciz = wmma16(af, load_b16(wiP, 64, F + n0, k0), ciz);
    cin = wmma16(af, load_b16(wiP, 64, 2 * F + n0, k0), cin);
    chr_ = wmma16(hf, load_b16(whP, 64, n0, k0), chr_);
    chz = wmma16(hf, load_b16(whP, 64, F + n0, k0), chz);
    chn = wmma16(hf, load_b16(whP, 64, 2 * F + n0, k0), chn);
  }
  const int lane = threadIdx.x & 31;
  const int hi = lane >> 4;
  const int n = n0 + (lane & 15);
#pragma unroll
  for (int r = 0; r < 8; ++r) {
    const int m = m0 + r + 8 * hi;
    if (m < Nn) {
      const float rg = sigmoid_f(cir[r] + bi[n] + chr_[r] + bh[n]);
      const float zg = sigmoid_f(ciz[r] + bi[F + n] + chz[r] + bh[F + n]);
      const float ng =
          tanhf(cin[r] + bi[2 * F + n] + rg * (chn[r] + bh[2 * F + n]));
      const float hold = (float)H[(size_t)m * 64 + n];
      Hn[(size_t)m * 64 + n] = (_Float16)((1.0f - zg) * ng + zg * hold);
    }
  }
}

// gated gather: out = sigmoid([h,feat] @ iw^T + ib) * (h @ jw^T + jb)
__global__ void k_gather(const _Float16* H, int F, const _Float16* X, int ldx,
                         int Kcpad, const _Float16* iP, const float* ib,
                         const _Float16* jP, const float* jb, _Float16* out16,
                         float* out32, int Nn, int Npad) {
  const int wid = threadIdx.x >> 5;
  const int m0 = (blockIdx.x * (blockDim.x >> 5) + wid) * 16;
  if (m0 >= Npad) return;
  const int n0 = blockIdx.y * 16;
  v8f ci = {0.f, 0.f, 0.f, 0.f, 0.f, 0.f, 0.f, 0.f};
  v8f cj = ci;
  for (int k0 = 0; k0 < Kcpad; k0 += 32) {
    v16h af = load_a16_cat(H, 64, F, X, ldx, m0, k0);
    ci = wmma16(af, load_b16(iP, Kcpad, n0, k0), ci);
  }
#pragma unroll
  for (int k0 = 0; k0 < 64; k0 += 32) {
    v16h af = load_a16(H, 64, m0, k0);
    cj = wmma16(af, load_b16(jP, 64, n0, k0), cj);
  }
  const int lane = threadIdx.x & 31;
  const int hi = lane >> 4;
  const int n = n0 + (lane & 15);
#pragma unroll
  for (int r = 0; r < 8; ++r) {
    const int m = m0 + r + 8 * hi;
    if (m < Nn) {
      const float val = sigmoid_f(ci[r] + ib[n]) * (cj[r] + jb[n]);
      if (out16) out16[(size_t)m * 64 + n] = (_Float16)val;
      if (out32) out32[(size_t)m * 64 + n] = val;
    }
  }
}

// ---------------- readout kernels ----------------

__global__ void k_cums(const int* bnn, int B, int* cum, int* ligcum) {
  if (blockIdx.x != 0 || threadIdx.x != 0) return;
  int c = 0;
  cum[0] = 0;
  for (int i = 0; i < B; ++i) { c += bnn[i]; cum[i + 1] = c; }
  int l = 0;
  for (int g = 0; g < B / 2; ++g) { ligcum[g] = l; l += bnn[2 * g]; }
  ligcum[B / 2] = l;
}

__global__ void k_pool(const float* h2, const int* cum, int B, float* pooled,
                       int Nn) {
  const int v = blockIdx.x;
  if (v >= Nn) return;
  int s = 0;
  while (s < B && v >= cum[s + 1]) ++s;
  if (s < B && (s & 1) == 0)
    atomicAdd(&pooled[(s >> 1) * 64 + threadIdx.x],
              h2[(long long)v * 64 + threadIdx.x]);
}

__global__ void k_fcnn(const float* pooled, const float* w1, const float* b1,
                       const float* w2, const float* b2, const float* ow,
                       const float* ob, float* xout, int G) {
  __shared__ float x1[32 * 128];
  __shared__ float x2[32 * 64];
  const int tid = threadIdx.x;
  for (int idx = tid; idx < G * 128; idx += blockDim.x) {
    const int g = idx >> 7, o = idx & 127;
    float a = b1[o];
    const float* p = pooled + g * 64;
    const float* w = w1 + o * 64;
    for (int k = 0; k < 64; ++k) a += p[k] * w[k];
    x1[idx] = fmaxf(a, 0.0f);
  }
  __syncthreads();
  for (int idx = tid; idx < G * 64; idx += blockDim.x) {
    const int g = idx >> 6, o = idx & 63;
    float a = b2[o];
    const float* w = w2 + o * 128;
    for (int k = 0; k < 128; ++k) a += x1[g * 128 + k] * w[k];
    x2[idx] = fmaxf(a, 0.0f);
  }
  __syncthreads();
  for (int idx = tid; idx < G; idx += blockDim.x) {
    float a = ob[0];
    for (int k = 0; k < 64; ++k) a += x2[idx * 64 + k] * ow[k];
    xout[idx] = a;
  }
}

__global__ void k_znorm(const float* h2, const int* cum, const int* ligcum, int B,
                        _Float16* Z, int Nn) {
  const int v = blockIdx.x;
  if (v >= Nn) return;
  int s = 0;
  while (s < B && v >= cum[s + 1]) ++s;
  if (s >= B || (s & 1)) return;
  const int i = ligcum[s >> 1] + (v - cum[s]);
  const int f = threadIdx.x;  // 64 threads
  const float val = h2[(long long)v * 64 + f];
  __shared__ float red[64];
  red[f] = val * val;
  __syncthreads();
  for (int off = 32; off > 0; off >>= 1) {
    if (f < off) red[f] += red[f + off];
    __syncthreads();
  }
  const float nrm = fmaxf(sqrtf(red[0]), 1e-12f);
  Z[(size_t)i * 64 + f] = (_Float16)(val / nrm);
}

// A_pred = sigmoid(Z @ Z^T), Z f16 [Lpad x 64]
__global__ void k_apred(const _Float16* Z, int L, int Lpad, float* A) {
  const int wid = threadIdx.x >> 5;
  const int m0 = (blockIdx.x * (blockDim.x >> 5) + wid) * 16;
  if (m0 >= Lpad) return;
  const int n0 = blockIdx.y * 16;
  v8f c = {0.f, 0.f, 0.f, 0.f, 0.f, 0.f, 0.f, 0.f};
#pragma unroll
  for (int k0 = 0; k0 < 64; k0 += 32) {
    v16h af = load_a16(Z, 64, m0, k0);
    v16h bf = load_b16(Z, 64, n0, k0);  // B[k][n] = Z[n][k]
    c = wmma16(af, bf, c);
  }
  const int lane = threadIdx.x & 31;
  const int hi = lane >> 4;
  const int n = n0 + (lane & 15);
  if (n < L) {
#pragma unroll
    for (int r = 0; r < 8; ++r) {
      const int m = m0 + r + 8 * hi;
      if (m < L) A[(long long)m * L + n] = sigmoid_f(c[r]);
    }
  }
}

// ---------------- host orchestration ----------------

extern "C" void kernel_launch(void* const* d_in, const int* in_sizes, int n_in,
                              void* d_out, int out_size, void* d_ws, size_t ws_size,
                              hipStream_t stream) {
  const float* feat = (const float*)d_in[0];
  const int* src1 = (const int*)d_in[1];
  const int* dst1 = (const int*)d_in[2];
  const int* et1 = (const int*)d_in[3];
  const int* src2 = (const int*)d_in[4];
  const int* dst2 = (const int*)d_in[5];
  const int* et2 = (const int*)d_in[6];
  const int* bnn = (const int*)d_in[7];
  const float* W_et1 = (const float*)d_in[9];
  const float* b_et1 = (const float*)d_in[10];
  const float* g1wi = (const float*)d_in[11];
  const float* g1wh = (const float*)d_in[12];
  const float* g1bi = (const float*)d_in[13];
  const float* g1bh = (const float*)d_in[14];
  const float* i1w = (const float*)d_in[15];
  const float* i1b = (const float*)d_in[16];
  const float* j1w = (const float*)d_in[17];
  const float* j1b = (const float*)d_in[18];
  const float* W_et2 = (const float*)d_in[19];
  const float* b_et2 = (const float*)d_in[20];
  const float* g2wi = (const float*)d_in[21];
  const float* g2wh = (const float*)d_in[22];
  const float* g2bi = (const float*)d_in[23];
  const float* g2bh = (const float*)d_in[24];
  const float* i2w = (const float*)d_in[25];
  const float* i2b = (const float*)d_in[26];
  const float* j2w = (const float*)d_in[27];
  const float* j2b = (const float*)d_in[28];
  const float* fc1w = (const float*)d_in[29];
  const float* fc1b = (const float*)d_in[30];
  const float* fc2w = (const float*)d_in[31];
  const float* fc2b = (const float*)d_in[32];
  const float* outw = (const float*)d_in[33];
  const float* outb = (const float*)d_in[34];
  (void)n_in;
  (void)ws_size;

  const int FIN = 44;
  const int N = in_sizes[0] / FIN;
  const int E1 = in_sizes[1];
  const int E2 = in_sizes[4];
  const int B = in_sizes[7];
  const int G = B / 2;
  const int Npad = (N + 15) & ~15;

  long long rem = (long long)out_size - G - (long long)N * 64;
  const int L = (rem > 0) ? (int)(std::sqrt((double)rem) + 0.5) : 0;
  const int Lpad = (L + 15) & ~15;

  float* xout = (float*)d_out;
  float* Aout = xout + G;
  float* h2out = Aout + (long long)L * L;

  char* wp = (char*)d_ws;
  auto carve = [&](size_t bytes) -> void* {
    void* p = (void*)wp;
    wp += (bytes + 255) & ~(size_t)255;
    return p;
  };
  _Float16* hA = (_Float16*)carve((size_t)Npad * 64 * 2);
  _Float16* hB = (_Float16*)carve((size_t)Npad * 64 * 2);
  _Float16* aP = (_Float16*)carve((size_t)Npad * 64 * 2);
  _Float16* h1gP = (_Float16*)carve((size_t)Npad * 64 * 2);
  _Float16* featP = (_Float16*)carve((size_t)Npad * 48 * 2);
  float* sB = (float*)carve((size_t)Npad * 256 * 4);
  float* cnt = (float*)carve((size_t)Npad * 8 * 4);
  _Float16* Zp = (_Float16*)carve((size_t)(Lpad > 0 ? Lpad : 16) * 64 * 2);
  float* pooled = (float*)carve(64 * 64 * 4);
  int* cum = (int*)carve(64 * 4);
  int* ligcum = (int*)carve(64 * 4);
  // packed f16 weights
  _Float16* wi1P = (_Float16*)carve(144 * 64 * 2);
  _Float16* wh1P = (_Float16*)carve(144 * 64 * 2);
  _Float16* wi2P = (_Float16*)carve(192 * 64 * 2);
  _Float16* wh2P = (_Float16*)carve(192 * 64 * 2);
  _Float16* i1P = (_Float16*)carve(64 * 96 * 2);
  _Float16* j1P = (_Float16*)carve(64 * 64 * 2);
  _Float16* i2P = (_Float16*)carve(64 * 128 * 2);
  _Float16* j2P = (_Float16*)carve(64 * 64 * 2);
  _Float16* et1P = (_Float16*)carve(48 * 256 * 2);
  _Float16* et2P = (_Float16*)carve(64 * 256 * 2);

  auto zero = [&](void* p, long long nfloats) {
    unsigned g = (unsigned)((nfloats + 255) / 256);
    k_zero<<<g, 256, 0, stream>>>((float*)p, nfloats);
  };
  auto pack = [&](const float* W, int rows, int K, _Float16* out, int Kpad) {
    const int tot = rows * Kpad;
    k_packB<<<(tot + 255) / 256, 256, 0, stream>>>(W, rows, K, out, Kpad);
  };

  // pack weights (cheap, once per launch)
  pack(g1wi, 144, 48, wi1P, 64);
  pack(g1wh, 144, 48, wh1P, 64);
  pack(g2wi, 192, 64, wi2P, 64);
  pack(g2wh, 192, 64, wh2P, 64);
  pack(i1w, 64, 92, i1P, 96);
  pack(j1w, 64, 48, j1P, 64);
  pack(i2w, 64, 128, i2P, 128);
  pack(j2w, 64, 64, j2P, 64);
  k_packEt<<<(48 * 256 + 255) / 256, 256, 0, stream>>>(W_et1, 48, 5, et1P, 256);
  k_packEt<<<(64 * 256 + 255) / 256, 256, 0, stream>>>(W_et2, 64, 4, et2P, 256);

  // zero pad-sensitive buffers (f16 buffers zeroed via f32 view)
  zero(hB, (long long)Npad * 32);
  zero(aP, (long long)Npad * 32);

  const unsigned mblocks = (unsigned)((Npad / 16 + 3) / 4);

  auto run_stage = [&](int F, int NE, int steps, const int* src, const int* dst,
                       const int* et, int E, const _Float16* etP,
                       const float* bet, const _Float16* wiP, const _Float16* whP,
                       const float* bi, const float* bh, const _Float16* iP,
                       const float* ib, const _Float16* jP, const float* jb,
                       const _Float16* xP, int ldx, int Kcpad, _Float16* out16,
                       float* out32) {
    _Float16* hc = hA;
    _Float16* hn = hB;
    const dim3 gm(mblocks, F / 16);
    for (int st = 0; st < steps; ++st) {
      zero(sB, (long long)Npad * 256);
      zero(cnt, (long long)Npad * NE);
      k_scatter<<<(unsigned)((E + 255) / 256), 256, 0, stream>>>(
          hc, F, src, dst, et, E, NE, sB, cnt);
      k_msg<<<gm, 128, 0, stream>>>(sB, cnt, etP, bet, aP, N, Npad, F, NE);
      k_gru<<<gm, 128, 0, stream>>>(aP, hc, wiP, whP, bi, bh, hn, N, Npad, F);
      _Float16* t = hc; hc = hn; hn = t;
    }
    const dim3 gg(mblocks, 4);
    k_gather<<<gg, 128, 0, stream>>>(hc, F, xP, ldx, Kcpad, iP, ib, jP, jb,
                                     out16, out32, N, Npad);
  };

  // stage 1 init: hA = pad(feat)->64 cols, featP = pad(feat)->48 cols
  {
    const long long t64 = (long long)Npad * 64;
    const long long t48 = (long long)Npad * 48;
    k_cvtpad<<<(unsigned)((t64 + 255) / 256), 256, 0, stream>>>(feat, N, FIN, hA,
                                                                64, Npad);
    k_cvtpad<<<(unsigned)((t48 + 255) / 256), 256, 0, stream>>>(feat, N, FIN,
                                                                featP, 48, Npad);
  }
  run_stage(48, 5, 2, src1, dst1, et1, E1, et1P, b_et1, wi1P, wh1P, g1bi, g1bh,
            i1P, i1b, j1P, j1b, featP, 48, 96, h1gP, nullptr);

  // stage 2 init: hA = copy(h1gP)
  k_copy16<<<(unsigned)(((long long)Npad * 64 + 255) / 256), 256, 0, stream>>>(
      h1gP, hA, (long long)Npad * 64);
  run_stage(64, 4, 2, src2, dst2, et2, E2, et2P, b_et2, wi2P, wh2P, g2bi, g2bh,
            i2P, i2b, j2P, j2b, h1gP, 64, 128, nullptr, h2out);

  // readout
  k_cums<<<1, 1, 0, stream>>>(bnn, B, cum, ligcum);
  zero(pooled, (long long)G * 64);
  k_pool<<<N, 64, 0, stream>>>(h2out, cum, B, pooled, N);
  k_fcnn<<<1, 256, 0, stream>>>(pooled, fc1w, fc1b, fc2w, fc2b, outw, outb, xout,
                                G);
  k_znorm<<<N, 64, 0, stream>>>(h2out, cum, ligcum, B, Zp, N);
  if (L > 0) {
    const dim3 ga((unsigned)((Lpad / 16 + 3) / 4), (unsigned)(Lpad / 16));
    k_apred<<<ga, 128, 0, stream>>>(Zp, L, Lpad, Aout);
  }
}
